// GTAM_3D_22196390986141
// MI455X (gfx1250) — compile-verified, matrix operand order checked by reading, not run on hardware
//
#include <hip/hip_runtime.h>
#include <hip/hip_bf16.h>
#include <cstdint>

#define NODES 20000
#define EDGES 640000
#define HD    128
#define GG    50
#define GP    64       // G padded to WMMA-friendly K
#define LL    6
#define CUTF  10.0f
#define NGR   64
#define SA    136      // LDS row stride in halves (128 + 8 pad)

typedef __attribute__((ext_vector_type(16))) _Float16 v16h;
typedef __attribute__((ext_vector_type(8)))  float    v8f;

__device__ __forceinline__ float sspf(float x) {
  // softplus(x) - ln(2), numerically stable
  float ax = fabsf(x);
  return fmaxf(x, 0.0f) + log1pf(__expf(-ax)) - 0.69314718056f;
}

// Async Global->LDS copy of 16 bytes per active lane (CDNA5 GLOBAL_LOAD_ASYNC_
// TO_LDS_B128, GV mode). LDS byte address = low 32 bits of the generic shared
// pointer (addrspacecast AS3->flat keeps the LDS offset in the low word).
// Tracked with ASYNCcnt; pair with s_wait_asynccnt 0 before the barrier.
__device__ __forceinline__ void async_b128(const _Float16* lds_dst, const _Float16* gsrc) {
  unsigned l = (unsigned)(size_t)lds_dst;
  unsigned long long g = (unsigned long long)(size_t)gsrc;
  asm volatile("global_load_async_to_lds_b128 %0, %1, off" :: "v"(l), "v"(g) : "memory");
}
__device__ __forceinline__ void wait_async0() {
  asm volatile("s_wait_asynccnt 0x0" ::: "memory");
}

// Load one 16x32 f16 WMMA operand fragment from LDS (row-major, stride in
// halves). lane&15 selects the A-row / B-column (B stored K-transposed so the
// same loader works); elements run over K per the CDNA5 16-bit A-matrix VGPR
// layout (CDNA5 ISA 7.12.2): lanes 16-31 are K+8, elements 8..15
// are K+16.
__device__ __forceinline__ v16h ld_frag(const _Float16* s, int idx0, int kk, int lane) {
  const _Float16* p = s + (idx0 + (lane & 15)) * SA + kk + ((lane >> 4) * 8);
  v16h r;
#pragma unroll
  for (int q = 0; q < 4; ++q) { r[2*q]   = p[2*q];      r[2*q+1] = p[2*q+1]; }
#pragma unroll
  for (int q = 0; q < 4; ++q) { r[8+2*q] = p[16+2*q];   r[9+2*q] = p[17+2*q]; }
  return r;
}

__device__ __forceinline__ v8f wmma_f16(v16h a, v16h b, v8f c) {
  return __builtin_amdgcn_wmma_f32_16x16x32_f16(false, a, false, b, (short)0, c, false, false);
}

// ---------------------------------------------------------------- utility ---
__global__ void k_fill0(float* p, size_t n) {
  size_t i = (size_t)blockIdx.x * blockDim.x + threadIdx.x;
  if (i < n) p[i] = 0.0f;
}

// f32 [l][K=128][N=128] -> f16 transposed [l][N=128][K=128]
__global__ void k_f32f16t(const float* s, _Float16* d, int layers) {
  int i = blockIdx.x * blockDim.x + threadIdx.x;
  if (i >= layers * HD * HD) return;
  int l = i / (HD * HD), r = i % (HD * HD);
  int k = r >> 7, n = r & 127;
  d[((size_t)l * HD + n) * HD + k] = (_Float16)s[i];
}

// mlp_w1 f32 [l][50][128] -> f16 transposed+padded [l][N=128][K=64]
__global__ void k_padw1t(const float* s, _Float16* d) {
  int i = blockIdx.x * blockDim.x + threadIdx.x;
  if (i >= LL * HD * GP) return;
  int l = i / (HD * GP), r = i % (HD * GP);
  int n = r >> 6, k = r & 63;
  d[i] = (k < GG) ? (_Float16)s[((size_t)l * GG + k) * HD + n] : (_Float16)0.0f;
}

__global__ void k_embed(const int* z, const float* emb, float* h) {
  int i = blockIdx.x * blockDim.x + threadIdx.x;
  if (i >= NODES * HD) return;
  int node = i >> 7;
  h[i] = emb[(size_t)z[node] * HD + (i & 127)];
}

// distance, cosine cutoff, Gaussian smearing -> f16 [E][64] (K padded)
__global__ void k_edge_geom(const float* pos, const int* ei, float* Cq, _Float16* eaG) {
  int e = blockIdx.x * blockDim.x + threadIdx.x;
  if (e >= EDGES) return;
  int s = ei[e], d = ei[EDGES + e];
  float dx = pos[s*3+0] - pos[d*3+0];
  float dy = pos[s*3+1] - pos[d*3+1];
  float dz = pos[s*3+2] - pos[d*3+2];
  float dist = sqrtf(dx*dx + dy*dy + dz*dz);
  Cq[e] = 0.5f * (__cosf(dist * 3.14159265358979f / CUTF) + 1.0f);
  const float step  = CUTF / (GG - 1);
  const float coeff = -0.5f / (step * step);
  _Float16* o = eaG + (size_t)e * GP;
  for (int g = 0; g < GP; ++g) {
    float t = dist - g * step;
    o[g] = (g < GG) ? (_Float16)__expf(coeff * t * t) : (_Float16)0.0f;
  }
}

// stage transposed weight [128 rows][K halves] into LDS via async copies;
// chunksPerRow = K/8 (16B chunks). All 256 lanes active (counts are multiples
// of 256), caller must wait_async0() + __syncthreads() before use.
__device__ __forceinline__ void stage_w_async(_Float16* sBt, const _Float16* Wt,
                                              int K, int tid) {
  int cpr = K >> 3;                       // chunks per row
  int total = HD * cpr;
  for (int idx = tid; idx < total; idx += 256) {
    int n = idx / cpr, c8 = (idx % cpr) * 8;
    async_b128(sBt + n * SA + c8, Wt + (size_t)n * K + c8);
  }
}

// ------------------------------------------------ single GEMM: xf = h @ W ---
__global__ __launch_bounds__(256) void k_node_gemm(const float* A, const _Float16* Wt,
                                                   float* outp, int M) {
  __shared__ alignas(16) _Float16 sA[64 * SA];
  __shared__ alignas(16) _Float16 sBt[HD * SA];
  int tid = threadIdx.x, lane = tid & 31, w = tid >> 5;
  int r0g = blockIdx.x * 64;
  stage_w_async(sBt, Wt, HD, tid);
  for (int idx = tid; idx < 64 * 32; idx += 256) {
    int r = idx >> 5, c4 = (idx & 31) * 4, gr = r0g + r;
    float4 v = (gr < M) ? *(const float4*)(A + (size_t)gr * HD + c4)
                        : make_float4(0.f, 0.f, 0.f, 0.f);
    _Float16* d = sA + r * SA + c4;
    d[0] = (_Float16)v.x; d[1] = (_Float16)v.y; d[2] = (_Float16)v.z; d[3] = (_Float16)v.w;
  }
  wait_async0();
  __syncthreads();
  int wm = w & 3, wn = w >> 2, r0 = wm * 16, n0 = wn * 64;
  v8f acc[4];
#pragma unroll
  for (int t = 0; t < 4; ++t)
#pragma unroll
    for (int i = 0; i < 8; ++i) acc[t][i] = 0.0f;
  for (int kk = 0; kk < HD; kk += 32) {
    v16h a = ld_frag(sA, r0, kk, lane);
#pragma unroll
    for (int t = 0; t < 4; ++t) {
      v16h b = ld_frag(sBt, n0 + t * 16, kk, lane);
      acc[t] = wmma_f16(a, b, acc[t]);
    }
  }
  int half = lane >> 4, cl = lane & 15;
#pragma unroll
  for (int t = 0; t < 4; ++t) {
    int c = n0 + t * 16 + cl;
#pragma unroll
    for (int r = 0; r < 8; ++r) {
      int gr = r0g + r0 + r + 8 * half;
      if (gr < M) outp[(size_t)gr * HD + c] = acc[t][r];
    }
  }
}

// ------------- fused two-GEMM node kernel: out = ssp(A@W1+b1)@W2+b2 (+res) ---
__global__ __launch_bounds__(256) void k_node_fused(const float* A,
                                                    const _Float16* W1t, const float* b1,
                                                    const _Float16* W2t, const float* b2,
                                                    const float* res, float* outp, int M) {
  __shared__ alignas(16) _Float16 sA[64 * SA];
  __shared__ alignas(16) _Float16 sBt[HD * SA];
  int tid = threadIdx.x, lane = tid & 31, w = tid >> 5;
  int r0g = blockIdx.x * 64;
  stage_w_async(sBt, W1t, HD, tid);
  for (int idx = tid; idx < 64 * 32; idx += 256) {
    int r = idx >> 5, c4 = (idx & 31) * 4, gr = r0g + r;
    float4 v = (gr < M) ? *(const float4*)(A + (size_t)gr * HD + c4)
                        : make_float4(0.f, 0.f, 0.f, 0.f);
    _Float16* d = sA + r * SA + c4;
    d[0] = (_Float16)v.x; d[1] = (_Float16)v.y; d[2] = (_Float16)v.z; d[3] = (_Float16)v.w;
  }
  wait_async0();
  __syncthreads();
  int wm = w & 3, wn = w >> 2, r0 = wm * 16, n0 = wn * 64;
  int half = lane >> 4, cl = lane & 15;
  v8f acc[4];
#pragma unroll
  for (int t = 0; t < 4; ++t)
#pragma unroll
    for (int i = 0; i < 8; ++i) acc[t][i] = 0.0f;
  for (int kk = 0; kk < HD; kk += 32) {
    v16h a = ld_frag(sA, r0, kk, lane);
#pragma unroll
    for (int t = 0; t < 4; ++t) {
      v16h b = ld_frag(sBt, n0 + t * 16, kk, lane);
      acc[t] = wmma_f16(a, b, acc[t]);
    }
  }
  __syncthreads();
  // restage W2 (async, overlaps the activation writes below)
  stage_w_async(sBt, W2t, HD, tid);
  // T1 = ssp(acc + b1) -> sA (f16)
#pragma unroll
  for (int t = 0; t < 4; ++t) {
    int c = n0 + t * 16 + cl;
    float bb = b1[c];
#pragma unroll
    for (int r = 0; r < 8; ++r) {
      int row = r0 + r + 8 * half;
      sA[row * SA + c] = (_Float16)sspf(acc[t][r] + bb);
    }
  }
  wait_async0();
  __syncthreads();
  v8f acc2[4];
#pragma unroll
  for (int t = 0; t < 4; ++t)
#pragma unroll
    for (int i = 0; i < 8; ++i) acc2[t][i] = 0.0f;
  for (int kk = 0; kk < HD; kk += 32) {
    v16h a = ld_frag(sA, r0, kk, lane);
#pragma unroll
    for (int t = 0; t < 4; ++t) {
      v16h b = ld_frag(sBt, n0 + t * 16, kk, lane);
      acc2[t] = wmma_f16(a, b, acc2[t]);
    }
  }
#pragma unroll
  for (int t = 0; t < 4; ++t) {
    int c = n0 + t * 16 + cl;
    float bb = b2[c];
#pragma unroll
    for (int r = 0; r < 8; ++r) {
      int gr = r0g + r0 + r + 8 * half;
      if (gr < M) {
        float v = acc2[t][r] + bb;
        if (res) v += res[(size_t)gr * HD + c];
        outp[(size_t)gr * HD + c] = v;
      }
    }
  }
}

// --- fused edge kernel: W = (ssp(ea@w1+b1)@w2+b2)*C ; agg[dst] += xf[src]*W ---
__global__ __launch_bounds__(256) void k_edge_msg(const _Float16* eaG, const float* Cq,
                                                  const int* ei,
                                                  const _Float16* w1t, const float* b1,
                                                  const _Float16* w2t, const float* b2,
                                                  const float* xf, float* agg) {
  __shared__ alignas(16) _Float16 sA[64 * SA];
  __shared__ alignas(16) _Float16 sBt[HD * SA];
  __shared__ int   sSrc[64];
  __shared__ int   sDst[64];
  __shared__ float sC[64];
  int tid = threadIdx.x, lane = tid & 31, w = tid >> 5;
  int e0 = blockIdx.x * 64;
  // edge_attr tile (f16, K padded to 64): async, 16B chunks, 512 chunks
  for (int idx = tid; idx < 64 * 8; idx += 256) {
    int r = idx >> 3, c = (idx & 7) * 8;
    async_b128(sA + r * SA + c, eaG + (size_t)(e0 + r) * GP + c);
  }
  stage_w_async(sBt, w1t, GP, tid);       // w1 transposed [128][64]
  if (tid < 64) {
    sSrc[tid] = ei[e0 + tid];
    sDst[tid] = ei[EDGES + e0 + tid];
    sC[tid]   = Cq[e0 + tid];
  }
  wait_async0();
  __syncthreads();
  int wm = w & 3, wn = w >> 2, r0 = wm * 16, n0 = wn * 64;
  int half = lane >> 4, cl = lane & 15;
  v8f acc[4];
#pragma unroll
  for (int t = 0; t < 4; ++t)
#pragma unroll
    for (int i = 0; i < 8; ++i) acc[t][i] = 0.0f;
  for (int kk = 0; kk < GP; kk += 32) {              // K = 64
    v16h a = ld_frag(sA, r0, kk, lane);
#pragma unroll
    for (int t = 0; t < 4; ++t) {
      v16h b = ld_frag(sBt, n0 + t * 16, kk, lane);
      acc[t] = wmma_f16(a, b, acc[t]);
    }
  }
  __syncthreads();
  stage_w_async(sBt, w2t, HD, tid);       // restage w2, overlaps ssp writes
#pragma unroll
  for (int t = 0; t < 4; ++t) {
    int c = n0 + t * 16 + cl;
    float bb = b1[c];
#pragma unroll
    for (int r = 0; r < 8; ++r) {
      int row = r0 + r + 8 * half;
      sA[row * SA + c] = (_Float16)sspf(acc[t][r] + bb);
    }
  }
  wait_async0();
  __syncthreads();
  v8f acc2[4];
#pragma unroll
  for (int t = 0; t < 4; ++t)
#pragma unroll
    for (int i = 0; i < 8; ++i) acc2[t][i] = 0.0f;
  for (int kk = 0; kk < HD; kk += 32) {              // K = 128
    v16h a = ld_frag(sA, r0, kk, lane);
#pragma unroll
    for (int t = 0; t < 4; ++t) {
      v16h b = ld_frag(sBt, n0 + t * 16, kk, lane);
      acc2[t] = wmma_f16(a, b, acc2[t]);
    }
  }
  // message + scatter-add, fused (never materialize W in HBM)
#pragma unroll
  for (int t = 0; t < 4; ++t) {
    int c = n0 + t * 16 + cl;
    float bb = b2[c];
#pragma unroll
    for (int r = 0; r < 8; ++r) {
      int row = r0 + r + 8 * half;
      float wv = (acc2[t][r] + bb) * sC[row];
      float xv = xf[(size_t)sSrc[row] * HD + c];
      atomicAdd(&agg[(size_t)sDst[row] * HD + c], wv * xv);
    }
  }
}

// ----------------------------------------------------------------- readout --
__global__ void k_readout(const float* hout, const int* batch, float* sums, float* cnt) {
  int i = blockIdx.x * blockDim.x + threadIdx.x;
  if (i >= NODES * HD) return;
  int node = i >> 7, c = i & 127, g = batch[node];
  atomicAdd(&sums[g * HD + c], hout[i]);
  if (c == 0) atomicAdd(&cnt[g], 1.0f);
}

__global__ void k_final(const float* sums, const float* cnt, float* outp) {
  int i = blockIdx.x * blockDim.x + threadIdx.x;
  if (i >= NGR * HD) return;
  outp[i] = sums[i] / fmaxf(cnt[i >> 7], 1.0f);
}

// ===========================================================================
extern "C" void kernel_launch(void* const* d_in, const int* in_sizes, int n_in,
                              void* d_out, int out_size, void* d_ws, size_t ws_size,
                              hipStream_t stream) {
  (void)in_sizes; (void)n_in; (void)out_size; (void)ws_size;
  const int*   z       = (const int*)  d_in[0];
  const float* pos     = (const float*)d_in[1];
  const int*   ei      = (const int*)  d_in[2];
  const int*   batch   = (const int*)  d_in[3];
  const float* emb     = (const float*)d_in[4];
  const float* mlp_w1  = (const float*)d_in[5];
  const float* mlp_b1  = (const float*)d_in[6];
  const float* mlp_w2  = (const float*)d_in[7];
  const float* mlp_b2  = (const float*)d_in[8];
  const float* conv_w1 = (const float*)d_in[9];
  const float* conv_w2 = (const float*)d_in[10];
  const float* conv_b2 = (const float*)d_in[11];
  const float* lin_w   = (const float*)d_in[12];
  const float* lin_b   = (const float*)d_in[13];
  const float* out1_w  = (const float*)d_in[14];
  const float* out1_b  = (const float*)d_in[15];
  const float* out2_w  = (const float*)d_in[16];
  const float* out2_b  = (const float*)d_in[17];
  float* outp = (float*)d_out;

  char* ws = (char*)d_ws;
  size_t off = 0;
  auto take = [&](size_t b) -> char* {
    char* p = ws + off;
    off = (off + b + 255) & ~(size_t)255;
    return p;
  };
  float*    h    = (float*)   take((size_t)NODES * HD * 4);
  float*    xf   = (float*)   take((size_t)NODES * HD * 4);
  float*    agg  = (float*)   take((size_t)NODES * HD * 4);
  float*    Cq   = (float*)   take((size_t)EDGES * 4);
  _Float16* eaG  = (_Float16*)take((size_t)EDGES * GP * 2);
  _Float16* w1t  = (_Float16*)take((size_t)LL * HD * GP * 2);   // [l][N][K=64]
  _Float16* w2t  = (_Float16*)take((size_t)LL * HD * HD * 2);   // [l][N][K]
  _Float16* cw1t = (_Float16*)take((size_t)LL * HD * HD * 2);
  _Float16* cw2t = (_Float16*)take((size_t)LL * HD * HD * 2);
  _Float16* lwt  = (_Float16*)take((size_t)LL * HD * HD * 2);
  _Float16* o1t  = (_Float16*)take((size_t)HD * HD * 2);
  _Float16* o2t  = (_Float16*)take((size_t)HD * HD * 2);
  float*    sums = (float*)   take((size_t)NGR * HD * 4);
  float*    cnt  = (float*)   take((size_t)NGR * 4);
  float*    hout = xf;  // xf is dead after the layer loop

  // one-time (per call) weight conversion f32 -> f16, K-transposed in global
  k_padw1t <<<(LL*HD*GP + 255)/256, 256, 0, stream>>>(mlp_w1, w1t);
  k_f32f16t<<<(LL*HD*HD + 255)/256, 256, 0, stream>>>(mlp_w2,  w2t,  LL);
  k_f32f16t<<<(LL*HD*HD + 255)/256, 256, 0, stream>>>(conv_w1, cw1t, LL);
  k_f32f16t<<<(LL*HD*HD + 255)/256, 256, 0, stream>>>(conv_w2, cw2t, LL);
  k_f32f16t<<<(LL*HD*HD + 255)/256, 256, 0, stream>>>(lin_w,   lwt,  LL);
  k_f32f16t<<<(HD*HD + 255)/256,    256, 0, stream>>>(out1_w,  o1t,  1);
  k_f32f16t<<<(HD*HD + 255)/256,    256, 0, stream>>>(out2_w,  o2t,  1);

  k_embed    <<<(NODES*HD + 255)/256, 256, 0, stream>>>(z, emb, h);
  k_edge_geom<<<(EDGES + 255)/256,    256, 0, stream>>>(pos, ei, Cq, eaG);

  const int nblk = (NODES + 63) / 64;
  for (int l = 0; l < LL; ++l) {
    k_fill0<<<(unsigned)(((size_t)NODES*HD + 255)/256), 256, 0, stream>>>(agg, (size_t)NODES*HD);
    k_node_gemm<<<nblk, 256, 0, stream>>>(h, cw1t + (size_t)l*HD*HD, xf, NODES);
    k_edge_msg <<<EDGES/64, 256, 0, stream>>>(eaG, Cq, ei,
                                              w1t + (size_t)l*HD*GP, mlp_b1 + l*HD,
                                              w2t + (size_t)l*HD*HD, mlp_b2 + l*HD,
                                              xf, agg);
    k_node_fused<<<nblk, 256, 0, stream>>>(agg,
                                           cw2t + (size_t)l*HD*HD, conv_b2 + l*HD,
                                           lwt  + (size_t)l*HD*HD, lin_b   + l*HD,
                                           h, h, NODES);
  }
  // output head: hout = ssp(h@out1+b1)@out2+b2
  k_node_fused<<<nblk, 256, 0, stream>>>(h, o1t, out1_b, o2t, out2_b,
                                         nullptr, hout, NODES);
  // segment mean readout
  k_fill0<<<(NGR*HD + 255)/256, 256, 0, stream>>>(sums, (size_t)NGR*HD);
  k_fill0<<<1, 256, 0, stream>>>(cnt, (size_t)NGR);
  k_readout<<<(NODES*HD + 255)/256, 256, 0, stream>>>(hout, batch, sums, cnt);
  k_final  <<<(NGR*HD + 255)/256,   256, 0, stream>>>(sums, cnt, outp);
}
